// AttModel_syb_40235253629201
// MI455X (gfx1250) — matrix operand order repaired and checked by reading, not compiled, not run on hardware
//
#include <hip/hip_runtime.h>
#include <math.h>

typedef __attribute__((ext_vector_type(16))) _Float16 v16h;
typedef __attribute__((ext_vector_type(8)))  _Float16 v8h;
typedef __attribute__((ext_vector_type(4)))  _Float16 v4h;
typedef __attribute__((ext_vector_type(8)))  float    v8f;

union H16 { v16h v; v8h q[2]; _Float16 h[16]; };
union F8  { v8f  v; float    f[8];  };

#define NEGMASK (-2147483648.0f)   /* jnp.float32(-2.0**31) */

// Swizzle a k-index within each 32-chunk so that a lane's 16 A-operand halves
// are contiguous: block order 0,1,2,3 -> 0,2,1,3 (2-bit block bit-reverse).
__device__ __forceinline__ int swz(int c) {
  int b  = (c >> 3) & 3;
  int pb = ((b & 1) << 1) | (b >> 1);
  return (c & ~31) | (c & 7) | (pb << 3);
}

__device__ __forceinline__ float4 f4zero() { return make_float4(0.f, 0.f, 0.f, 0.f); }

// ---------------------------------------------------------------------------
// Generic WMMA GEMM:  C = act( A[gather] @ W + bias (+ addRow) )
//   Block tile 128x64, 4 waves (128 thr); each wave owns 32x64 (8 wmma/step).
//   K-step 32, f16 staging in LDS. Staging is branchless and load-batched so
//   all 12 global_load_b128 per K-step are in flight before any wait.
// ---------------------------------------------------------------------------
__global__ __launch_bounds__(128)
void gemm_wmma(const float* __restrict__ A, const int* __restrict__ gidx,
               const float* __restrict__ W, const float* __restrict__ bias,
               const float* __restrict__ addRow, float* __restrict__ C,
               int M, int K, int N, int lda,
               int seg, long long segStride, int ldc, int act)
{
  __shared__ _Float16 As[128][40];   // [m][swz(k)]
  __shared__ _Float16 Bst[64][40];   // [n][k]
  const int tid  = threadIdx.x;
  const int w    = tid >> 5;
  const int lane = tid & 31;
  const int half = lane >> 4;
  const int l16  = lane & 15;
  const int bm   = blockIdx.x * 128;
  const int bn   = blockIdx.y * 64;

  // hoisted per-thread A row bases (constant across K loop); invalid rows
  // clamp to row 0 and are zero-selected after the load.
  long long abase[8];
  bool      aval[8];
  const int arsub = tid >> 3;        // 0..15
  const int ac4   = (tid & 7) * 4;   // 0,4,..,28
  #pragma unroll
  for (int i = 0; i < 8; i++) {
    int ar = bm + i * 16 + arsub;
    aval[i] = ar < M;
    int arc = aval[i] ? ar : 0;
    long long arow = gidx ? (long long)gidx[arc] : (long long)arc;
    abase[i] = arow * (long long)lda;
  }

  F8 acc[2][4];
  #pragma unroll
  for (int m = 0; m < 2; m++)
    #pragma unroll
    for (int t = 0; t < 4; t++)
      #pragma unroll
      for (int i = 0; i < 8; i++) acc[m][t].f[i] = 0.0f;

  for (int k0 = 0; k0 < K; k0 += 32) {
    const int  ac  = k0 + ac4;
    const bool kok = ac < K;               // K % 4 == 0 at every call site
    const long long acl = kok ? ac : 0;

    // ---- batched loads: A tile 128x32 + W tile 32x64, no interleaved waits
    float4 va[8];
    #pragma unroll
    for (int i = 0; i < 8; i++)
      va[i] = *(const float4*)&A[abase[i] + acl];
    float4 vb[4];
    #pragma unroll
    for (int i = 0; i < 4; i++) {
      int e  = i * 128 + tid;
      int r  = e >> 4;                     // k row
      int c4 = (e & 15) * 4;               // n
      int kr = k0 + r;
      vb[i] = *(const float4*)&W[(long long)(kr < K ? kr : 0) * N + bn + c4];
    }
    if (k0 + 32 + lane < K)
      __builtin_prefetch(&W[(long long)(k0 + 32 + lane) * N + bn + w * 16], 0, 3);

    // ---- convert + store to LDS (zero-select out-of-range) ----
    #pragma unroll
    for (int i = 0; i < 8; i++) {
      float4 v = (aval[i] && kok) ? va[i] : f4zero();
      v4h hv;
      hv[0] = (_Float16)v.x; hv[1] = (_Float16)v.y;
      hv[2] = (_Float16)v.z; hv[3] = (_Float16)v.w;
      *(v4h*)&As[i * 16 + arsub][swz(ac4)] = hv;
    }
    #pragma unroll
    for (int i = 0; i < 4; i++) {
      int e  = i * 128 + tid;
      int r  = e >> 4;
      int c4 = (e & 15) * 4;
      float4 v = (k0 + r < K) ? vb[i] : f4zero();
      Bst[c4 + 0][r] = (_Float16)v.x;      // transposed [n][k]
      Bst[c4 + 1][r] = (_Float16)v.y;
      Bst[c4 + 2][r] = (_Float16)v.z;
      Bst[c4 + 3][r] = (_Float16)v.w;
    }
    __syncthreads();

    H16 bmat[4];
    #pragma unroll
    for (int t = 0; t < 4; t++) {
      bmat[t].q[0] = *(const v8h*)&Bst[t * 16 + l16][half * 8];
      bmat[t].q[1] = *(const v8h*)&Bst[t * 16 + l16][16 + half * 8];
    }
    #pragma unroll
    for (int ms = 0; ms < 2; ms++) {
      H16 a;
      a.q[0] = *(const v8h*)&As[w * 32 + ms * 16 + l16][half * 16];
      a.q[1] = *(const v8h*)&As[w * 32 + ms * 16 + l16][half * 16 + 8];
      #pragma unroll
      for (int t = 0; t < 4; t++)
        acc[ms][t].v = __builtin_amdgcn_wmma_f32_16x16x32_f16(
            false, a.v, false, bmat[t].v, (short)0, acc[ms][t].v, false, false);
    }
    __syncthreads();
  }

  #pragma unroll
  for (int ms = 0; ms < 2; ms++)
    #pragma unroll
    for (int t = 0; t < 4; t++)
      #pragma unroll
      for (int r = 0; r < 8; r++) {
        int mm = bm + w * 32 + ms * 16 + r + half * 8;
        int nn = bn + t * 16 + l16;
        if (mm < M) {
          float v = acc[ms][t].f[r] + bias[nn];
          if (addRow) v += addRow[(long long)(mm % seg) * N + nn];
          if (act) v = fmaxf(v, 0.0f);
          long long off = (long long)(mm / seg) * segStride
                        + (long long)(mm % seg) * ldc + nn;
          C[off] = v;
        }
      }
}

// ---------------------------------------------------------------------------
// Flash-style masked attention (per block: 64 queries, one (b,h) pair).
//   gmode: 0 none | 1 cross-modal blocks | 2 intra-modal graphs. LS=512 LQ=64.
//   Qs/Ps stored in swizzled A-operand layout; Vst transposed (d x k); Ks
//   row-major (already B-operand contiguous for Q@K^T). Staging load-batched.
// ---------------------------------------------------------------------------
__global__ __launch_bounds__(128)
void attn_wmma(const float* __restrict__ Qp, const float* __restrict__ Kp,
               const float* __restrict__ Vp, const float* __restrict__ keymask,
               const float* __restrict__ qmask,
               const float* __restrict__ sybMaskG, const float* __restrict__ qMaskG,
               const float* __restrict__ sybGraph, const float* __restrict__ qGraph,
               float* __restrict__ Out, int Tq, int Tk, int gmode)
{
  __shared__ _Float16 Qs[64][72];    // [q][swz(d)]
  __shared__ _Float16 Ks[64][72];    // [k][d]
  __shared__ _Float16 Vst[64][72];   // [d][k]
  __shared__ _Float16 Ps[64][72];    // [q][swz(k)]
  __shared__ float    kmS[64];

  const int tid  = threadIdx.x;
  const int w    = tid >> 5;
  const int lane = tid & 31;
  const int half = lane >> 4;
  const int l16  = lane & 15;
  const int q0   = blockIdx.x * 64;
  const int h    = blockIdx.y;
  const int b    = blockIdx.z;
  const int HH   = 512;

  // stage Q tile (clamped loads, zero-select past Tq), swizzled
  #pragma unroll
  for (int ph = 0; ph < 2; ph++) {
    float4 vq[4];
    #pragma unroll
    for (int i = 0; i < 4; i++) {
      int e  = (ph * 4 + i) * 128 + tid;
      int r  = e >> 4;
      int c4 = (e & 15) * 4;
      int qc = q0 + r; if (qc >= Tq) qc = q0;   // q0 < Tq always
      vq[i] = *(const float4*)&Qp[((long long)b * Tq + qc) * HH + h * 64 + c4];
    }
    #pragma unroll
    for (int i = 0; i < 4; i++) {
      int e  = (ph * 4 + i) * 128 + tid;
      int r  = e >> 4;
      int c4 = (e & 15) * 4;
      float4 v = (q0 + r < Tq) ? vq[i] : f4zero();
      v4h hv;
      hv[0] = (_Float16)v.x; hv[1] = (_Float16)v.y;
      hv[2] = (_Float16)v.z; hv[3] = (_Float16)v.w;
      *(v4h*)&Qs[r][swz(c4)] = hv;
    }
  }

  F8 oacc[4];
  #pragma unroll
  for (int t = 0; t < 4; t++)
    #pragma unroll
    for (int i = 0; i < 8; i++) oacc[t].f[i] = 0.0f;
  float mrow[8], lrow[8];
  #pragma unroll
  for (int r = 0; r < 8; r++) { mrow[r] = -__builtin_inff(); lrow[r] = 0.0f; }

  const int nkt = Tk >> 6;           // Tk = 576 -> 9 tiles
  for (int kt = 0; kt < nkt; kt++) {
    __syncthreads();                 // protect Ks/Vst/Ps reuse (covers Qs too)
    const int k0 = kt * 64;
    #pragma unroll
    for (int ph = 0; ph < 2; ph++) {
      float4 kv[4], vv[4];
      #pragma unroll
      for (int i = 0; i < 4; i++) {
        int e  = (ph * 4 + i) * 128 + tid;
        int r  = e >> 4;
        int c4 = (e & 15) * 4;
        long long src = ((long long)b * Tk + k0 + r) * HH + h * 64 + c4;
        kv[i] = *(const float4*)&Kp[src];
        vv[i] = *(const float4*)&Vp[src];
      }
      #pragma unroll
      for (int i = 0; i < 4; i++) {
        int e  = (ph * 4 + i) * 128 + tid;
        int r  = e >> 4;
        int c4 = (e & 15) * 4;
        v4h hk;
        hk[0] = (_Float16)kv[i].x; hk[1] = (_Float16)kv[i].y;
        hk[2] = (_Float16)kv[i].z; hk[3] = (_Float16)kv[i].w;
        *(v4h*)&Ks[r][c4] = hk;          // row-major: B-operand contiguous
        Vst[c4 + 0][r] = (_Float16)vv[i].x;  // transposed
        Vst[c4 + 1][r] = (_Float16)vv[i].y;
        Vst[c4 + 2][r] = (_Float16)vv[i].z;
        Vst[c4 + 3][r] = (_Float16)vv[i].w;
      }
    }
    if (tid < 64) kmS[tid] = keymask[(long long)b * Tk + k0 + tid];
    __syncthreads();

    // ---- scores: S = Q @ K^T ----
    F8 sAcc[4];
    #pragma unroll
    for (int t = 0; t < 4; t++)
      #pragma unroll
      for (int i = 0; i < 8; i++) sAcc[t].f[i] = 0.0f;
    #pragma unroll
    for (int c2 = 0; c2 < 2; c2++) {
      H16 a;
      a.q[0] = *(const v8h*)&Qs[w * 16 + l16][c2 * 32 + half * 16];
      a.q[1] = *(const v8h*)&Qs[w * 16 + l16][c2 * 32 + half * 16 + 8];
      #pragma unroll
      for (int t = 0; t < 4; t++) {
        H16 bb;
        bb.q[0] = *(const v8h*)&Ks[t * 16 + l16][c2 * 32 + half * 8];
        bb.q[1] = *(const v8h*)&Ks[t * 16 + l16][c2 * 32 + 16 + half * 8];
        sAcc[t].v = __builtin_amdgcn_wmma_f32_16x16x32_f16(
            false, a.v, false, bb.v, (short)0, sAcc[t].v, false, false);
      }
    }

    // ---- scale + mask ----
    #pragma unroll
    for (int t = 0; t < 4; t++)
      #pragma unroll
      for (int r = 0; r < 8; r++) {
        float sv = sAcc[t].f[r] * 0.125f;  // 1/sqrt(64)
        int q = q0 + w * 16 + r + half * 8;
        int k = k0 + t * 16 + l16;
        bool pass = (q < Tq) && (kmS[t * 16 + l16] > 0.0f);
        if (pass && gmode == 1) {            // cross-modal off-diagonal only
          float g;
          if (q < 512 && k >= 512)      g = qMaskG[(long long)b * 64 + (k - 512)];
          else if (q >= 512 && k < 512) g = sybMaskG[(long long)b * 512 + k];
          else                          g = 0.0f;
          pass = g > 0.0f;
        } else if (pass && gmode == 2) {     // intra-modal structured graphs
          float g;
          if (q < 512 && k < 512)
            g = sybGraph[(long long)b * 512 * 512 + (long long)q * 512 + k];
          else if (q >= 512 && k >= 512)
            g = qGraph[(long long)b * 64 * 64 + (long long)(q - 512) * 64 + (k - 512)];
          else g = 1.0f;
          pass = g > 0.0f;
        }
        sAcc[t].f[r] = pass ? sv : NEGMASK;
      }

    // ---- online softmax update (finite NEG => uniform on fully-masked) ----
    #pragma unroll
    for (int r = 0; r < 8; r++) {
      float tm = fmaxf(fmaxf(sAcc[0].f[r], sAcc[1].f[r]),
                       fmaxf(sAcc[2].f[r], sAcc[3].f[r]));
      tm = fmaxf(tm, __shfl_xor(tm, 1));
      tm = fmaxf(tm, __shfl_xor(tm, 2));
      tm = fmaxf(tm, __shfl_xor(tm, 4));
      tm = fmaxf(tm, __shfl_xor(tm, 8));
      float newm = fmaxf(mrow[r], tm);
      float c = expf(mrow[r] - newm);
      mrow[r] = newm;
      lrow[r] *= c;
      #pragma unroll
      for (int t = 0; t < 4; t++) oacc[t].f[r] *= c;
      float ls = 0.0f;
      #pragma unroll
      for (int t = 0; t < 4; t++) {
        float p = expf(sAcc[t].f[r] - newm);
        ls += p;
        Ps[w * 16 + r + half * 8][swz(t * 16 + l16)] = (_Float16)p;
      }
      ls += __shfl_xor(ls, 1);
      ls += __shfl_xor(ls, 2);
      ls += __shfl_xor(ls, 4);
      ls += __shfl_xor(ls, 8);
      lrow[r] += ls;
    }
    __syncthreads();  // Ps visible to all waves

    // ---- out += P @ V ----
    #pragma unroll
    for (int c2 = 0; c2 < 2; c2++) {
      H16 a;
      a.q[0] = *(const v8h*)&Ps[w * 16 + l16][c2 * 32 + half * 16];
      a.q[1] = *(const v8h*)&Ps[w * 16 + l16][c2 * 32 + half * 16 + 8];
      #pragma unroll
      for (int dt = 0; dt < 4; dt++) {
        H16 bb;
        bb.q[0] = *(const v8h*)&Vst[dt * 16 + l16][c2 * 32 + half * 8];
        bb.q[1] = *(const v8h*)&Vst[dt * 16 + l16][c2 * 32 + 16 + half * 8];
        oacc[dt].v = __builtin_amdgcn_wmma_f32_16x16x32_f16(
            false, a.v, false, bb.v, (short)0, oacc[dt].v, false, false);
      }
    }
  }

  #pragma unroll
  for (int dt = 0; dt < 4; dt++)
    #pragma unroll
    for (int r = 0; r < 8; r++) {
      int q = q0 + w * 16 + r + half * 8;
      if (q < Tq) {
        float v = (oacc[dt].f[r] / lrow[r]) * qmask[(long long)b * Tq + q];
        Out[((long long)b * Tq + q) * HH + h * 64 + dt * 16 + l16] = v;
      }
    }
}

// ---------------------------------------------------------------------------
// LayerNorm( A + B ), H=512, wave-per-row (wave32), 8 rows / block
// ---------------------------------------------------------------------------
__global__ __launch_bounds__(256)
void ln_add(const float* __restrict__ A, const float* __restrict__ Bres,
            const float* __restrict__ g, const float* __restrict__ bb,
            float* __restrict__ O, int rows)
{
  int row  = blockIdx.x * 8 + (threadIdx.x >> 5);
  int lane = threadIdx.x & 31;
  if (row >= rows) return;
  long long base = (long long)row * 512;
  float v[16];
  float s = 0.0f;
  #pragma unroll
  for (int j = 0; j < 4; j++) {
    float4 a4 = *(const float4*)&A[base + lane * 4 + j * 128];
    float4 b4 = *(const float4*)&Bres[base + lane * 4 + j * 128];
    v[j * 4 + 0] = a4.x + b4.x; v[j * 4 + 1] = a4.y + b4.y;
    v[j * 4 + 2] = a4.z + b4.z; v[j * 4 + 3] = a4.w + b4.w;
    s += v[j * 4 + 0] + v[j * 4 + 1] + v[j * 4 + 2] + v[j * 4 + 3];
  }
  for (int d = 1; d < 32; d <<= 1) s += __shfl_xor(s, d);
  float mean = s * (1.0f / 512.0f);
  float vs = 0.0f;
  #pragma unroll
  for (int j = 0; j < 16; j++) { float t = v[j] - mean; vs += t * t; }
  for (int d = 1; d < 32; d <<= 1) vs += __shfl_xor(vs, d);
  float rstd = rsqrtf(vs * (1.0f / 512.0f) + 1e-6f);
  #pragma unroll
  for (int j = 0; j < 4; j++) {
    int c = lane * 4 + j * 128;
    float4 g4 = *(const float4*)&g[c];
    float4 bb4 = *(const float4*)&bb[c];
    float4 o4;
    o4.x = g4.x * (v[j * 4 + 0] - mean) * rstd + bb4.x;
    o4.y = g4.y * (v[j * 4 + 1] - mean) * rstd + bb4.y;
    o4.z = g4.z * (v[j * 4 + 2] - mean) * rstd + bb4.z;
    o4.w = g4.w * (v[j * 4 + 3] - mean) * rstd + bb4.w;
    *(float4*)&O[base + c] = o4;
  }
}

// qmask[row] = sign(sum |x_row|)   (H = 512)
__global__ __launch_bounds__(256)
void rowmask_kernel(const float* __restrict__ X, float* __restrict__ Qm, int rows)
{
  int row  = blockIdx.x * 8 + (threadIdx.x >> 5);
  int lane = threadIdx.x & 31;
  if (row >= rows) return;
  long long base = (long long)row * 512;
  float s = 0.0f;
  #pragma unroll
  for (int j = 0; j < 4; j++) {
    float4 a4 = *(const float4*)&X[base + lane * 4 + j * 128];
    s += fabsf(a4.x) + fabsf(a4.y) + fabsf(a4.z) + fabsf(a4.w);
  }
  for (int d = 1; d < 32; d <<= 1) s += __shfl_xor(s, d);
  if (lane == 0) Qm[row] = (s > 0.0f) ? 1.0f : 0.0f;
}

// mask = concat(syb_mask, q_mask) over S=576
__global__ void concat_mask(const float* __restrict__ sm, const float* __restrict__ qm,
                            float* __restrict__ m)
{
  int i = blockIdx.x * 256 + threadIdx.x;
  if (i < 16 * 576) {
    int b = i / 576, s = i % 576;
    m[i] = (s < 512) ? sm[b * 512 + s] : qm[b * 64 + (s - 512)];
  }
}

// dec = dec_tab[1] * sqrt(512), broadcast over B=16
__global__ void dec_init(const float* __restrict__ dec_tab, float* __restrict__ dec)
{
  int i = blockIdx.x * 256 + threadIdx.x;
  if (i < 16 * 512) dec[i] = dec_tab[512 + (i & 511)] * 22.62741699796952f;
}

// ---------------------------------------------------------------------------
extern "C" void kernel_launch(void* const* d_in, const int* in_sizes, int n_in,
                              void* d_out, int out_size, void* d_ws, size_t ws_size,
                              hipStream_t stream)
{
  (void)in_sizes; (void)n_in; (void)out_size; (void)ws_size;
  const int B = 16, LS = 512, LQ = 64, S = 576, H = 512, NB = 6;
  const int NR  = B * S * H;          // 4,718,592 floats
  const int SEGB = 1 << 30;           // "no segmentation"

  const int*   syb_ipt   = (const int*)  d_in[0];
  const float* syb_mask  = (const float*)d_in[1];
  const float* syb_graph = (const float*)d_in[2];
  const float* q_fea     = (const float*)d_in[3];
  const float* q_graph   = (const float*)d_in[4];
  const float* q_maskI   = (const float*)d_in[5];
  const float* emb_table = (const float*)d_in[6];
  const float* sW1 = (const float*)d_in[7],  *sb1 = (const float*)d_in[8];
  const float* sW2 = (const float*)d_in[9],  *sb2 = (const float*)d_in[10];
  const float* syb_pos = (const float*)d_in[11];
  const float* qW1 = (const float*)d_in[12], *qb1 = (const float*)d_in[13];
  const float* qW2 = (const float*)d_in[14], *qb2 = (const float*)d_in[15];
  const float* q_pos   = (const float*)d_in[16];
  const float* dec_tab = (const float*)d_in[17];
  const float* eWq = (const float*)d_in[18], *ebq = (const float*)d_in[19];
  const float* eWk = (const float*)d_in[20], *ebk = (const float*)d_in[21];
  const float* eWv = (const float*)d_in[22], *ebv = (const float*)d_in[23];
  const float* eg  = (const float*)d_in[24], *ebp = (const float*)d_in[25];
  const float* efW1= (const float*)d_in[26], *efb1= (const float*)d_in[27];
  const float* efW2= (const float*)d_in[28], *efb2= (const float*)d_in[29];
  const float* efg = (const float*)d_in[30], *efb = (const float*)d_in[31];
  const float* dWq = (const float*)d_in[32], *dbq = (const float*)d_in[33];
  const float* dWk = (const float*)d_in[34], *dbk = (const float*)d_in[35];
  const float* dWv = (const float*)d_in[36], *dbv = (const float*)d_in[37];
  const float* dg  = (const float*)d_in[38], *dbp = (const float*)d_in[39];
  const float* dfW1= (const float*)d_in[40], *dfb1= (const float*)d_in[41];
  const float* dfW2= (const float*)d_in[42], *dfb2= (const float*)d_in[43];
  const float* dfg = (const float*)d_in[44], *dfb = (const float*)d_in[45];
  float* out = (float*)d_out;

  // workspace arena (floats)
  float* x    = (float*)d_ws;            // B*S*H
  float* x2   = x    + NR;
  float* tmp  = x2   + NR;               // attn-out / ffn-out / q-hidden
  float* hid  = tmp  + NR;               // B*S*2048 : also holds Qp/Kp/Vp
  float* msk  = hid  + (long long)B * S * 2048;
  float* qm   = msk  + B * S;
  float* dec  = qm   + B * S;            // 16x512
  float* dec2 = dec  + B * H;
  float* dtmp = dec2 + B * H;
  float* dhid = dtmp + B * H;            // 16x2048
  float* dq   = dhid + B * 2048;
  float* qmd  = dq   + B * H;            // 16
  float* Qp = hid, *Kp = hid + NR, *Vp = hid + 2 * NR;

  dim3 blk(128);

  // ---- embedding path: syb = relu(emb[syb]@sW1+sb1)@sW2+sb2 + syb_pos ----
  gemm_wmma<<<dim3(64, 32), blk, 0, stream>>>(emb_table, syb_ipt, sW1, sb1, nullptr,
      hid, B * LS, 300, 2048, 300, SEGB, 0, 2048, 1);
  gemm_wmma<<<dim3(64, 8), blk, 0, stream>>>(hid, nullptr, sW2, sb2, syb_pos,
      x, B * LS, 2048, H, 2048, LS, (long long)S * H, H, 0);
  // ---- q path: q = relu(q_fea@qW1+qb1)@qW2+qb2 + q_pos ----
  gemm_wmma<<<dim3(8, 8), blk, 0, stream>>>(q_fea, nullptr, qW1, qb1, nullptr,
      tmp, B * LQ, 300, H, 300, SEGB, 0, H, 1);
  gemm_wmma<<<dim3(8, 8), blk, 0, stream>>>(tmp, nullptr, qW2, qb2, q_pos,
      x + (long long)LS * H, B * LQ, H, H, H, LQ, (long long)S * H, H, 0);

  concat_mask<<<36, 256, 0, stream>>>(syb_mask, q_maskI, msk);

  // ---- encoder blocks ----
  for (int i = 0; i < NB; i++) {
    int gmode = (i < 2) ? 0 : (i < 4) ? 1 : 2;
    const long long wO = (long long)i * H * H, bO = (long long)i * H;
    const long long f1O = (long long)i * H * 2048, f1b = (long long)i * 2048;
    rowmask_kernel<<<1152, 256, 0, stream>>>(x, qm, B * S);
    gemm_wmma<<<dim3(72, 8), blk, 0, stream>>>(x, nullptr, eWq + wO, ebq + bO,
        nullptr, Qp, B * S, H, H, H, SEGB, 0, H, 1);
    gemm_wmma<<<dim3(72, 8), blk, 0, stream>>>(x, nullptr, eWk + wO, ebk + bO,
        nullptr, Kp, B * S, H, H, H, SEGB, 0, H, 1);
    gemm_wmma<<<dim3(72, 8), blk, 0, stream>>>(x, nullptr, eWv + wO, ebv + bO,
        nullptr, Vp, B * S, H, H, H, SEGB, 0, H, 1);
    attn_wmma<<<dim3(9, 8, 16), blk, 0, stream>>>(Qp, Kp, Vp, msk, qm,
        syb_mask, q_maskI, syb_graph, q_graph, tmp, S, S, gmode);
    ln_add<<<1152, 256, 0, stream>>>(tmp, x, eg + bO, ebp + bO, x2, B * S);
    gemm_wmma<<<dim3(72, 32), blk, 0, stream>>>(x2, nullptr, efW1 + f1O, efb1 + f1b,
        nullptr, hid, B * S, H, 2048, H, SEGB, 0, 2048, 1);
    gemm_wmma<<<dim3(72, 8), blk, 0, stream>>>(hid, nullptr, efW2 + f1O, efb2 + bO,
        nullptr, tmp, B * S, 2048, H, 2048, SEGB, 0, H, 0);
    ln_add<<<1152, 256, 0, stream>>>(tmp, x2, efg + bO, efb + bO, x, B * S);
  }

  // ---- decoder ----
  dec_init<<<32, 256, 0, stream>>>(dec_tab, dec);
  for (int i = 0; i < NB; i++) {
    const long long wO = (long long)i * H * H, bO = (long long)i * H;
    const long long f1O = (long long)i * H * 2048, f1b = (long long)i * 2048;
    rowmask_kernel<<<2, 256, 0, stream>>>(dec, qmd, B);
    gemm_wmma<<<dim3(1, 8), blk, 0, stream>>>(dec, nullptr, dWq + wO, dbq + bO,
        nullptr, dq, B, H, H, H, SEGB, 0, H, 1);
    gemm_wmma<<<dim3(72, 8), blk, 0, stream>>>(x, nullptr, dWk + wO, dbk + bO,
        nullptr, Kp, B * S, H, H, H, SEGB, 0, H, 1);
    gemm_wmma<<<dim3(72, 8), blk, 0, stream>>>(x, nullptr, dWv + wO, dbv + bO,
        nullptr, Vp, B * S, H, H, H, SEGB, 0, H, 1);
    attn_wmma<<<dim3(1, 8, 16), blk, 0, stream>>>(dq, Kp, Vp, msk, qmd,
        nullptr, nullptr, nullptr, nullptr, dtmp, 1, S, 0);
    ln_add<<<2, 256, 0, stream>>>(dtmp, dec, dg + bO, dbp + bO, dec2, B);
    gemm_wmma<<<dim3(1, 32), blk, 0, stream>>>(dec2, nullptr, dfW1 + f1O, dfb1 + f1b,
        nullptr, dhid, B, H, 2048, H, SEGB, 0, 2048, 1);
    gemm_wmma<<<dim3(1, 8), blk, 0, stream>>>(dhid, nullptr, dfW2 + f1O, dfb2 + bO,
        nullptr, dtmp, B, 2048, H, 2048, SEGB, 0, H, 0);
    ln_add<<<2, 256, 0, stream>>>(dtmp, dec2, dfg + bO, dfb + bO, dec, B);
  }

  hipMemcpyAsync(out, dec, (size_t)B * H * sizeof(float),
                 hipMemcpyDeviceToDevice, stream);
}
